// LSTM_41927470744005
// MI455X (gfx1250) — matrix operand order repaired and checked by reading, not compiled, not run on hardware
//
#include <hip/hip_runtime.h>
#include <hip/hip_bf16.h>
#include <stdint.h>

// ---------------------------------------------------------------------------
// LSTM on MI455X (gfx1250):
//   Phase 1: gates_x[t, 4096] = x_t @ Wx(1024x4096) + b   (dense WMMA GEMM)
//   Phase 2: persistent cooperative kernel, 32 WGs x 256 thr, global barrier
//            per step; recurrent GEMVs done as replicated-row WMMA (bf16).
// Weights are pre-packed into WMMA B-fragment order so the inner loops load
// 32B/lane contiguously (global_load_b128 pairs). h/c vectors are async-
// staged into LDS (global_load_async_to_lds_b128) once per step.
// ---------------------------------------------------------------------------

typedef __attribute__((ext_vector_type(16))) __bf16 v16bf;
typedef __attribute__((ext_vector_type(8)))  float  v8f;

#define T_STEPS   8192
#define HID       1024
#define G4        4096
#define ZIN       2048
#define NWG       32          // workgroups in the recurrence kernel
#define KC        (HID / 32)  // 32 k-chunks of 32

// ---- workspace layout (bytes) ---------------------------------------------
static constexpr size_t OFF_BAR = 0;                              // 2 uints
static constexpr size_t OFF_HBF = 256;                            // 1024 bf16
static constexpr size_t OFF_CBF = OFF_HBF + 2048;                 // 1024 bf16
static constexpr size_t OFF_CF  = OFF_CBF + 2048;                 // 1024 f32
static constexpr size_t OFF_BC  = OFF_CF  + 4096;                 // 4096 f32 gate bias
static constexpr size_t OFF_XBF = OFF_BC  + 16384;                // 8192*1024 bf16
static constexpr size_t OFF_WX  = OFF_XBF + (size_t)T_STEPS*HID*2;// packed 1024x4096
static constexpr size_t OFF_WH  = OFF_WX  + (size_t)HID*G4*2;     // packed 1024x4096
static constexpr size_t OFF_W4  = OFF_WH  + (size_t)HID*G4*2;     // packed 1024x1024
static constexpr size_t OFF_WO  = OFF_W4  + (size_t)HID*HID*2;    // packed 1024x1024
static constexpr size_t OFF_GX  = OFF_WO  + (size_t)HID*HID*2;    // 8192*4096 bf16

// ---- bf16 <-> f32 bit helpers (storage is raw uint16 bits) -----------------
__device__ __forceinline__ unsigned short f2bfu(float f) {
  unsigned u = __builtin_bit_cast(unsigned, f);
  unsigned r = u + 0x7FFFu + ((u >> 16) & 1u);   // round-to-nearest-even
  return (unsigned short)(r >> 16);
}
__device__ __forceinline__ float bfu2f(unsigned short s) {
  return __builtin_bit_cast(float, (unsigned)s << 16);
}
__device__ __forceinline__ float sigm(float x) {
  return 1.0f / (1.0f + __expf(-x));
}

// A-fragment: two contiguous 16B halves at p and p+16 (ushort units)
__device__ __forceinline__ v16bf ld_afrag(const unsigned short* p) {
  union { uint4 q[2]; v16bf v; } u;
  u.q[0] = *(const uint4*)(p);
  u.q[1] = *(const uint4*)(p + 16);
  return u.v;
}
// B-fragment: packed layout, 32B contiguous per lane
__device__ __forceinline__ v16bf ld_bfrag(const unsigned short* p) {
  return *(const v16bf*)(p);
}

// ---- async global->LDS staging (gfx1250 ASYNC path) ------------------------
// copies 2048 bytes (1024 bf16) using 4 x (32 lanes x b128)
__device__ __forceinline__ void stage_vec_async(unsigned short* lds_dst,
                                                const unsigned short* gsrc,
                                                int lane) {
  unsigned base = (unsigned)(size_t)lds_dst;   // low 32 bits = LDS byte offset
#pragma unroll
  for (int i = 0; i < 4; ++i) {
    unsigned loff = base + (unsigned)(i * 512 + lane * 16);
    unsigned long long g =
        (unsigned long long)(size_t)(gsrc + (size_t)i * 256 + (size_t)lane * 8);
    asm volatile("global_load_async_to_lds_b128 %0, %1, off"
                 :: "v"(loff), "v"(g) : "memory");
  }
  asm volatile("s_wait_asynccnt 0x0" ::: "memory");
}

// ---- device-wide sense barrier --------------------------------------------
__device__ __forceinline__ void gbarrier(unsigned* bar) {
  __syncthreads();
  if (threadIdx.x == 0) {
    __threadfence();
    unsigned* cnt = bar;
    unsigned* gen = bar + 1;
    unsigned g = __hip_atomic_load(gen, __ATOMIC_RELAXED, __HIP_MEMORY_SCOPE_AGENT);
    if (atomicAdd(cnt, 1u) == (unsigned)(NWG - 1)) {
      __hip_atomic_store(cnt, 0u, __ATOMIC_RELAXED, __HIP_MEMORY_SCOPE_AGENT);
      __hip_atomic_store(gen, g + 1u, __ATOMIC_RELEASE, __HIP_MEMORY_SCOPE_AGENT);
    } else {
      while (__hip_atomic_load(gen, __ATOMIC_ACQUIRE, __HIP_MEMORY_SCOPE_AGENT) == g)
        __builtin_amdgcn_s_sleep(2);
    }
    __threadfence();
  }
  __syncthreads();
}

// ===========================================================================
// Setup kernels
// ===========================================================================
__global__ void k_init_state(unsigned* bar, unsigned short* hbf,
                             unsigned short* cbf, float* cf) {
  int i = threadIdx.x;
  if (i < 2) bar[i] = 0u;
  for (int u = i; u < HID; u += blockDim.x) {
    hbf[u] = 0;  cbf[u] = 0;  cf[u] = 0.0f;
  }
}

__global__ void k_cvt_x(const float* __restrict__ x, unsigned short* __restrict__ xb,
                        int n) {
  for (int i = blockIdx.x * blockDim.x + threadIdx.x; i < n;
       i += gridDim.x * blockDim.x)
    xb[i] = f2bfu(x[i]);
}

// Pack w0..w3 [2048,1024] into WMMA B-fragment order.
// Packed index: i = ((tn*KC + kc)*32 + lane)*16 + e
//   k   = kc*32 + (e<8 ? 8h+e : 16+8h+e-8), h=lane>>4   (table order)
//   col = tn*16 + (lane&15)                   (gate column; gate g = col>>10)
// wxp gets z-rows 0..1023 (x part), whp gets z-rows 1024..2047 (h part).
__global__ void k_pack_gates(const float* __restrict__ w0, const float* __restrict__ w1,
                             const float* __restrict__ w2, const float* __restrict__ w3,
                             unsigned short* __restrict__ wxp,
                             unsigned short* __restrict__ whp) {
  const size_t n = (size_t)HID * G4;
  for (size_t i = blockIdx.x * blockDim.x + threadIdx.x; i < n;
       i += (size_t)gridDim.x * blockDim.x) {
    int e    = (int)(i & 15);
    int lane = (int)((i >> 4) & 31);
    int kc   = (int)((i >> 9) & 31);
    int tn   = (int)(i >> 14);                 // 0..255
    int k    = kc * 32 + ((e < 8) ? (lane >> 4) * 8 + e : 16 + (lane >> 4) * 8 + (e - 8));
    int col  = tn * 16 + (lane & 15);
    int g    = col >> 10;
    int u    = col & (HID - 1);
    const float* w = (g == 0) ? w0 : (g == 1) ? w1 : (g == 2) ? w2 : w3;
    wxp[i] = f2bfu(w[(size_t)k * HID + u]);
    whp[i] = f2bfu(w[(size_t)(k + HID) * HID + u]);
  }
}

// Pack w4 / wo [1024,1024] into the same fragment order (tn in 0..63).
__global__ void k_pack_sq(const float* __restrict__ w4, const float* __restrict__ wo,
                          unsigned short* __restrict__ w4p,
                          unsigned short* __restrict__ wop) {
  const size_t n = (size_t)HID * HID;
  for (size_t i = blockIdx.x * blockDim.x + threadIdx.x; i < n;
       i += (size_t)gridDim.x * blockDim.x) {
    int e    = (int)(i & 15);
    int lane = (int)((i >> 4) & 31);
    int kc   = (int)((i >> 9) & 31);
    int tn   = (int)(i >> 14);                 // 0..63
    int k    = kc * 32 + ((e < 8) ? (lane >> 4) * 8 + e : 16 + (lane >> 4) * 8 + (e - 8));
    int col  = tn * 16 + (lane & 15);
    w4p[i] = f2bfu(w4[(size_t)k * HID + col]);
    wop[i] = f2bfu(wo[(size_t)k * HID + col]);
  }
}

__global__ void k_bias(const float* __restrict__ b0, const float* __restrict__ b1,
                       const float* __restrict__ b2, const float* __restrict__ b3,
                       float* __restrict__ bc) {
  int col = blockIdx.x * blockDim.x + threadIdx.x;
  if (col < G4) {
    int g = col >> 10, u = col & (HID - 1);
    const float* b = (g == 0) ? b0 : (g == 1) ? b1 : (g == 2) ? b2 : b3;
    bc[col] = b[u];
  }
}

// ===========================================================================
// Phase 1: gates_x = X @ Wx + b   (8192x1024 @ 1024x4096), bf16 WMMA, f32 acc
// one 16x16 output tile per wave; K=1024 in 32 chunks of 32
// ===========================================================================
__global__ __launch_bounds__(256) void k_gemm_x(
    const unsigned short* __restrict__ xb, const unsigned short* __restrict__ wxp,
    const float* __restrict__ bc, unsigned short* __restrict__ gx) {
  const int wid  = threadIdx.x >> 5;
  const int lane = threadIdx.x & 31;
  const int lcol = lane & 15;
  const bool hi  = lane >= 16;

  const int tile = blockIdx.x * 8 + wid;   // 131072 tiles
  const int tm = tile >> 8;                // 0..511
  const int tn = tile & 255;               // 0..255
  const int row0 = tm * 16;

  v8f acc;
  {
    float bi = bc[tn * 16 + lcol];
#pragma unroll
    for (int v = 0; v < 8; ++v) acc[v] = bi;
  }

  const unsigned short* xrow = xb + (size_t)(row0 + lcol) * HID + (hi ? 8 : 0);
  const unsigned short* bp   = wxp + (((size_t)tn * KC) * 32 + lane) * 16;

#pragma unroll 4
  for (int kb = 0; kb < HID; kb += 32) {
    v16bf A = ld_afrag(xrow + kb);
    v16bf B = ld_bfrag(bp);
    bp += 512;                                   // next k-chunk block
    __builtin_prefetch(bp + 512, 0, 1);          // global_prefetch_b8
    acc = __builtin_amdgcn_wmma_f32_16x16x32_bf16(false, A, false, B,
                                                  (short)0, acc, false, false);
  }

  // D layout: acc[v] -> (M = v + (hi?8:0), N = lane&15)
  const int rbase = row0 + (hi ? 8 : 0);
#pragma unroll
  for (int v = 0; v < 8; ++v)
    gx[(size_t)(rbase + v) * G4 + tn * 16 + lcol] = f2bfu(acc[v]);
}

// ===========================================================================
// Phase 2: sequential recurrence.  NWG workgroups x 256 threads (8 wave32).
// WG b owns units [b*32, b*32+32).  Per step:
//   G: gates_h slice (8 tiles of 16 cols, 1 per wave), A = h replicated (LDS)
//   E: elementwise c update (32 lanes)
//   S: h = o * tanh(c @ w4 + b4) slice (2 tiles, waves 0..1), A = c (LDS)
//   O: out_t = h @ wo + bo slice (2 tiles, waves 0..1), overlaps next G
// Global barriers: after c-write and after h-write.
// h/c broadcast vectors async-staged into LDS once per step by wave 0.
// ===========================================================================
__global__ __launch_bounds__(256, 1) void k_lstm(
    const unsigned short* __restrict__ gx, const unsigned short* __restrict__ whp,
    const unsigned short* __restrict__ w4p, const unsigned short* __restrict__ wop,
    unsigned short* __restrict__ hbf, unsigned short* __restrict__ cbf,
    float* __restrict__ cf, unsigned* __restrict__ bar,
    const float* __restrict__ b4, const float* __restrict__ bo,
    float* __restrict__ out) {
  __shared__ unsigned short hs[HID];   // staged h_{t-1} / h_t
  __shared__ unsigned short cs[HID];   // staged c_t
  __shared__ float gbuf[4][32];
  __shared__ float o_l[32];

  const int wid  = threadIdx.x >> 5;
  const int lane = threadIdx.x & 31;
  const int lcol = lane & 15;
  const bool hi  = lane >= 16;
  const int u0   = blockIdx.x * 32;
  const int ha   = hi ? 8 : 0;

  // stage initial h (zeros)
  if (wid == 0) stage_vec_async(hs, hbf, lane);
  __syncthreads();

  for (int t = 0; t < T_STEPS; ++t) {
    // ---------------- Phase G: gate slice, one tile per wave ----------------
    {
      const int g      = wid >> 1;
      const int coloff = (wid & 1) * 16;
      const int gtile  = (g * HID + u0 + coloff) >> 4;       // tile index in 4096
      v8f acc;
      {
        float ci = bfu2f(gx[(size_t)t * G4 + gtile * 16 + lcol]); // x-part + bias
#pragma unroll
        for (int v = 0; v < 8; ++v) acc[v] = ci;
      }
      const unsigned short* bp = whp + (((size_t)gtile * KC) * 32 + lane) * 16;
#pragma unroll 4
      for (int kb = 0; kb < HID; kb += 32) {
        v16bf A = ld_afrag(&hs[kb + ha]);        // h replicated into all rows
        v16bf B = ld_bfrag(bp);
        bp += 512;
        acc = __builtin_amdgcn_wmma_f32_16x16x32_bf16(false, A, false, B,
                                                      (short)0, acc, false, false);
      }
      if (!hi) gbuf[g][coloff + lane] = acc[0];              // row 0 of D
    }
    __syncthreads();

    // ---------------- Phase E: elementwise cell update ----------------------
    if (threadIdx.x < 32) {
      const int lu = threadIdx.x;
      float f  = sigm(gbuf[0][lu]);
      float ii = sigm(gbuf[1][lu]);
      float gg = tanhf(gbuf[2][lu]);
      float oo = sigm(gbuf[3][lu]);
      float cn = f * cf[u0 + lu] + ii * gg;
      cf[u0 + lu]  = cn;
      cbf[u0 + lu] = f2bfu(cn);
      o_l[lu] = oo;
    }
    gbarrier(bar);   // all c slices visible
    if (wid == 0) stage_vec_async(cs, cbf, lane);
    __syncthreads();

    // ---------------- Phase S: h = o * tanh(c @ w4 + b4) --------------------
    if (wid < 2) {
      const int coloff = wid * 16;
      const int tile   = (u0 + coloff) >> 4;                 // tile in 1024
      v8f acc;
      {
        float ci = b4[tile * 16 + lcol];
#pragma unroll
        for (int v = 0; v < 8; ++v) acc[v] = ci;
      }
      const unsigned short* bp = w4p + (((size_t)tile * KC) * 32 + lane) * 16;
#pragma unroll 4
      for (int kb = 0; kb < HID; kb += 32) {
        v16bf A = ld_afrag(&cs[kb + ha]);
        v16bf B = ld_bfrag(bp);
        bp += 512;
        acc = __builtin_amdgcn_wmma_f32_16x16x32_bf16(false, A, false, B,
                                                      (short)0, acc, false, false);
      }
      if (!hi) {
        float hv = o_l[coloff + lane] * tanhf(acc[0]);
        hbf[u0 + coloff + lane] = f2bfu(hv);
      }
    }
    gbarrier(bar);   // all h slices visible
    if (wid == 0) stage_vec_async(hs, hbf, lane);
    __syncthreads();

    // ---------------- Phase O: out_t = h @ wo + bo (overlaps next G) --------
    if (wid < 2) {
      const int coloff = wid * 16;
      const int tile   = (u0 + coloff) >> 4;
      v8f acc;
      {
        float ci = bo[tile * 16 + lcol];
#pragma unroll
        for (int v = 0; v < 8; ++v) acc[v] = ci;
      }
      const unsigned short* bp = wop + (((size_t)tile * KC) * 32 + lane) * 16;
#pragma unroll 4
      for (int kb = 0; kb < HID; kb += 32) {
        v16bf A = ld_afrag(&hs[kb + ha]);
        v16bf B = ld_bfrag(bp);
        bp += 512;
        acc = __builtin_amdgcn_wmma_f32_16x16x32_bf16(false, A, false, B,
                                                      (short)0, acc, false, false);
      }
      if (!hi) out[(size_t)t * HID + u0 + coloff + lane] = acc[0];
    }
    // next-step Phase G reads hs (already staged + synced); no extra barrier
  }
}

// ===========================================================================
// host launcher
// ===========================================================================
extern "C" void kernel_launch(void* const* d_in, const int* in_sizes, int n_in,
                              void* d_out, int out_size, void* d_ws, size_t ws_size,
                              hipStream_t stream) {
  (void)in_sizes; (void)n_in; (void)out_size; (void)ws_size;

  const float* x  = (const float*)d_in[0];
  const float* w0 = (const float*)d_in[1];
  const float* b0 = (const float*)d_in[2];
  const float* w1 = (const float*)d_in[3];
  const float* b1 = (const float*)d_in[4];
  const float* w2 = (const float*)d_in[5];
  const float* b2 = (const float*)d_in[6];
  const float* w3 = (const float*)d_in[7];
  const float* b3 = (const float*)d_in[8];
  const float* w4 = (const float*)d_in[9];
  const float* b4 = (const float*)d_in[10];
  const float* wo = (const float*)d_in[11];
  const float* bo = (const float*)d_in[12];
  float* out = (float*)d_out;

  char* ws = (char*)d_ws;
  unsigned*       bar = (unsigned*)(ws + OFF_BAR);
  unsigned short* hbf = (unsigned short*)(ws + OFF_HBF);
  unsigned short* cbf = (unsigned short*)(ws + OFF_CBF);
  float*          cf  = (float*)(ws + OFF_CF);
  float*          bc  = (float*)(ws + OFF_BC);
  unsigned short* xbf = (unsigned short*)(ws + OFF_XBF);
  unsigned short* wxp = (unsigned short*)(ws + OFF_WX);
  unsigned short* whp = (unsigned short*)(ws + OFF_WH);
  unsigned short* w4p = (unsigned short*)(ws + OFF_W4);
  unsigned short* wop = (unsigned short*)(ws + OFF_WO);
  unsigned short* gx  = (unsigned short*)(ws + OFF_GX);

  k_init_state<<<1, 256, 0, stream>>>(bar, hbf, cbf, cf);
  k_cvt_x<<<4096, 256, 0, stream>>>(x, xbf, T_STEPS * HID);
  k_pack_gates<<<4096, 256, 0, stream>>>(w0, w1, w2, w3, wxp, whp);
  k_pack_sq<<<2048, 256, 0, stream>>>(w4, wo, w4p, wop);
  k_bias<<<16, 256, 0, stream>>>(b0, b1, b2, b3, bc);

  // 512 x 256 = 131072 tiles, 8 tiles (waves) per 256-thread block
  k_gemm_x<<<16384, 256, 0, stream>>>(xbf, wxp, bc, gx);

  // persistent cooperative recurrence
  k_lstm<<<NWG, 256, 0, stream>>>(gx, whp, w4p, wop, hbf, cbf, cf, bar,
                                  b4, bo, out);
}
